// MeshDeformerDecoder_7344394076591
// MI455X (gfx1250) — matrix operand order repaired and checked by reading, not compiled, not run on hardware
//
#include <hip/hip_runtime.h>
#include <cstdint>
#include <cstddef>

// ---------- problem constants ----------
#define B_      256
#define L_      256      // z columns
#define SRC_    1024     // c / hidden columns
#define INDIM   1280     // L_ + SRC_
#define NOUT    1024
#define NEXP    16
#define KTOT    (NEXP * INDIM)      // 20480
#define KSPLIT  8
#define KCHUNK  (KTOT / KSPLIT)     // 2560  (= exactly 2 experts)
#define KSTEPS  (KCHUNK / 32)       // 80

typedef __attribute__((ext_vector_type(16))) __bf16        v16bf;
typedef __attribute__((ext_vector_type(2)))  __bf16        v2bf;
typedef __attribute__((ext_vector_type(8)))  float         v8f;
typedef __attribute__((ext_vector_type(4)))  float         v4f;
typedef __attribute__((ext_vector_type(4)))  unsigned int  v4u;
typedef __attribute__((ext_vector_type(8)))  unsigned short v8us;

__device__ __forceinline__ unsigned short f2bf(float f) {
    unsigned u = __builtin_bit_cast(unsigned, f);
    u += 0x7FFFu + ((u >> 16) & 1u);          // round-to-nearest-even
    return (unsigned short)(u >> 16);
}

// pack 8 floats -> 8 bf16 (one v8us); V_CVT_PK_BF16_F32 if the toolchain has it
__device__ __forceinline__ v8us pack8(const float* f) {
#if __has_builtin(__builtin_amdgcn_cvt_pk_bf16_f32)
    union { v2bf p[4]; v8us v; } u;
#pragma unroll
    for (int j = 0; j < 4; ++j)
        u.p[j] = __builtin_amdgcn_cvt_pk_bf16_f32(f[2 * j], f[2 * j + 1]);
    return u.v;
#else
    v8us v;
#pragma unroll
    for (int j = 0; j < 8; ++j) v[j] = f2bf(f[j]);
    return v;
#endif
}

__device__ __forceinline__ float elu(float x) {
    return x > 0.f ? x : (__expf(x) - 1.f);
}

// ---------------------------------------------------------------------------
// Gating MLP: coeff = softmax(elu(elu(x@gw0+gb0)@gw1+gb1)@gw2+gb2)
// one block per token, 64 threads
// ---------------------------------------------------------------------------
__global__ __launch_bounds__(64)
void gating_kernel(const float* __restrict__ z, const float* __restrict__ c,
                   const float* __restrict__ gw0, const float* __restrict__ gb0,
                   const float* __restrict__ gw1, const float* __restrict__ gb1,
                   const float* __restrict__ gw2, const float* __restrict__ gb2,
                   float* __restrict__ coeff)
{
    __shared__ float x[INDIM];
    __shared__ float h0[64];
    __shared__ float h1[64];
    __shared__ float s[16];
    const int b = blockIdx.x;
    const int t = threadIdx.x;

    for (int i = t; i < L_;   i += 64) x[i]      = z[b * L_   + i];
    for (int i = t; i < SRC_; i += 64) x[L_ + i] = c[b * SRC_ + i];
    __syncthreads();

    float acc = gb0[t];
    for (int i = 0; i < INDIM; ++i) acc += x[i] * gw0[i * 64 + t];
    h0[t] = elu(acc);
    __syncthreads();

    acc = gb1[t];
    for (int i = 0; i < 64; ++i) acc += h0[i] * gw1[i * 64 + t];
    h1[t] = elu(acc);
    __syncthreads();

    if (t < 16) {
        float a2 = gb2[t];
        for (int i = 0; i < 64; ++i) a2 += h1[i] * gw2[i * 16 + t];
        s[t] = a2;
    }
    __syncthreads();

    if (t < 16) {
        float mx = s[0];
        for (int i = 1; i < 16; ++i) mx = fmaxf(mx, s[i]);
        float sum = 0.f;
        for (int i = 0; i < 16; ++i) sum += __expf(s[i] - mx);
        coeff[b * 16 + t] = __expf(s[t] - mx) / sum;
    }
}

// ---------------------------------------------------------------------------
// Expert GEMM (bf16 WMMA, split-K):
//   partial[ks][b][o] = sum_{k in split} coeff[b,k/1280]*inp[b,k%1280] * W[k][o]
// grid (8 nblk, 4 mblk, 8 ksplit), 256 threads (8 waves), wave tile 16x64
// ---------------------------------------------------------------------------
__global__ __launch_bounds__(256)
void expert_gemm(const float* __restrict__ z, const float* __restrict__ src,
                 const float* __restrict__ coeff, const float* __restrict__ W,
                 float* __restrict__ partial)
{
    // LDS row stride padded to 40 halfs (80B, 16B-aligned rows)
    __shared__ __align__(16) unsigned short ldsA[2][64 * 40];
    __shared__ __align__(16) unsigned short ldsB[2][128 * 40];

    const int tid   = threadIdx.x;
    const int nbase = blockIdx.x * 128;
    const int mbase = blockIdx.y * 64;
    const int kbase = blockIdx.z * KCHUNK;

    // ---- A-tile loader mapping: 64 rows x 32 halfs, 8 halfs per thread ----
    const int ar  = tid >> 2;          // row 0..63
    const int ac8 = (tid & 3) * 8;     // col base 0/8/16/24
    const int am  = mbase + ar;

    // ---- B-tile loader mapping: 32 k x 128 n, 16 elems per thread ----
    const int bn  = tid & 127;         // n 0..127
    const int bkq = tid >> 7;          // k-half 0/1 (16 rows each)

    // This k-split covers exactly experts e0 and e0+1 -> hoist coeff loads,
    // and e/i come from one compare instead of a division.
    const int   e0  = 2 * blockIdx.z;
    const float cs0 = coeff[am * NEXP + e0];
    const float cs1 = coeff[am * NEXP + e0 + 1];
    const float* pz = z   + (size_t)am * L_;
    const float* ps = src + (size_t)am * SRC_ - L_;   // ps+i valid for i>=L_

    float astage[8];
    float bstage[16];

    auto loadA = [&](int klocal) {                 // klocal in [0, KCHUNK)
        const int  klo = klocal + ac8;
        const bool hi  = klo >= INDIM;
        const int  i   = hi ? (klo - INDIM) : klo; // chunk never straddles 256
        const float cs = hi ? cs1 : cs0;
        const float* p = ((i < L_) ? pz : ps) + i;
        v4f lo = *(const v4f*)p;
        v4f hh = *(const v4f*)(p + 4);
#pragma unroll
        for (int j = 0; j < 4; ++j) { astage[j] = lo[j] * cs; astage[4 + j] = hh[j] * cs; }
    };
    auto storeA = [&](int buf) {
        *(v8us*)&ldsA[buf][ar * 40 + ac8] = pack8(astage);
    };
    auto loadB = [&](int k) {                      // k = global row in W
        const float* p = W + (size_t)(k + bkq * 16) * NOUT + nbase + bn;
#pragma unroll
        for (int j = 0; j < 16; ++j) bstage[j] = p[(size_t)j * NOUT];
    };
    auto storeB = [&](int buf) {
        *(v8us*)&ldsB[buf][bn * 40 + bkq * 16]     = pack8(bstage);
        *(v8us*)&ldsB[buf][bn * 40 + bkq * 16 + 8] = pack8(bstage + 8);
    };

    // ---- WMMA mapping (wave32) ----
    const int wave = tid >> 5;
    const int lane = tid & 31;
    const int wm   = wave & 3;          // m sub-tile (16 rows each)
    const int wn   = wave >> 2;         // 0..1 (64 cols each)
    const int lm   = lane & 15;
    const int half = lane >> 4;

    v8f acc[4] = {};

    // A frag: m=lm, kb = half*8; elems j0..7 -> k kb..kb+7, j8..15 -> 16+kb..
    const int aoff = (wm * 16 + lm) * 40 + half * 8;
    const int boff0 = (wn * 64 + lm) * 40 + half * 8;    // nt adds nt*16*40

    auto computeStep = [&](int buf) {
        // Batch ALL fragment loads first so the compiler emits one ds clause +
        // a single s_wait_dscnt, then issues the 4 WMMAs back-to-back
        // (independent accumulators -> no inter-WMMA hazards).
        union { v4u u[2]; v16bf v; } ca;
        union { v4u u[2]; v16bf v; } cb[4];
        ca.u[0] = *(const v4u*)&ldsA[buf][aoff];
        ca.u[1] = *(const v4u*)&ldsA[buf][aoff + 16];
#pragma unroll
        for (int nt = 0; nt < 4; ++nt) {
            const int boff = boff0 + nt * (16 * 40);
            cb[nt].u[0] = *(const v4u*)&ldsB[buf][boff];
            cb[nt].u[1] = *(const v4u*)&ldsB[buf][boff + 16];
        }
#pragma unroll
        for (int nt = 0; nt < 4; ++nt)
            acc[nt] = __builtin_amdgcn_wmma_f32_16x16x32_bf16(
                false, ca.v, false, cb[nt].v, (short)0, acc[nt], false, false);
    };

    // ---- double-buffered main loop over K ----
    loadA(0); loadB(kbase);
    storeA(0); storeB(0);
    __syncthreads();

    for (int ks = 0; ks < KSTEPS; ++ks) {
        const int cur = ks & 1;
        if (ks + 1 < KSTEPS) { loadA((ks + 1) * 32); loadB(kbase + (ks + 1) * 32); }
        computeStep(cur);
        __syncthreads();
        if (ks + 1 < KSTEPS) { storeA(cur ^ 1); storeB(cur ^ 1); }
        __syncthreads();
    }

    // ---- epilogue: deterministic per-split partial store ----
    float* pout = partial + (size_t)blockIdx.z * (B_ * NOUT);
#pragma unroll
    for (int nt = 0; nt < 4; ++nt) {
        const int n = nbase + wn * 64 + nt * 16 + lm;
#pragma unroll
        for (int r = 0; r < 8; ++r) {
            const int m = mbase + wm * 16 + r + half * 8;
            pout[(size_t)m * NOUT + n] = acc[nt][r];
        }
    }
}

// ---------------------------------------------------------------------------
// Reduce k-splits + coeff@bias + optional ELU
// ---------------------------------------------------------------------------
__global__ __launch_bounds__(256)
void reduce_act(const float* __restrict__ partial, const float* __restrict__ coeff,
                const float* __restrict__ bias, float* __restrict__ dst, int apply_elu)
{
    const int idx = blockIdx.x * 256 + threadIdx.x;   // 0 .. 256*1024-1
    const int bb  = idx >> 10;
    const int o   = idx & 1023;
    float acc = 0.f;
#pragma unroll
    for (int zt = 0; zt < KSPLIT; ++zt) acc += partial[(size_t)zt * (B_ * NOUT) + idx];
    const float* cf = coeff + bb * 16;
#pragma unroll
    for (int e = 0; e < 16; ++e) acc += cf[e] * bias[e * NOUT + o];
    dst[idx] = apply_elu ? elu(acc) : acc;
}

// ---------------------------------------------------------------------------
extern "C" void kernel_launch(void* const* d_in, const int* in_sizes, int n_in,
                              void* d_out, int out_size, void* d_ws, size_t ws_size,
                              hipStream_t stream) {
    const float* z   = (const float*)d_in[0];
    const float* c   = (const float*)d_in[1];
    const float* w0  = (const float*)d_in[2];
    const float* b0  = (const float*)d_in[3];
    const float* w1  = (const float*)d_in[4];
    const float* b1  = (const float*)d_in[5];
    const float* w2  = (const float*)d_in[6];
    const float* b2  = (const float*)d_in[7];
    const float* gw0 = (const float*)d_in[8];
    const float* gb0 = (const float*)d_in[9];
    const float* gw1 = (const float*)d_in[10];
    const float* gb1 = (const float*)d_in[11];
    const float* gw2 = (const float*)d_in[12];
    const float* gb2 = (const float*)d_in[13];
    float* out = (float*)d_out;

    char* ws = (char*)d_ws;
    float* coeff   = (float*)ws;                                   // 16 KB
    float* h       = (float*)(ws + 16384);                         // 1 MB
    float* partial = (float*)(ws + 16384 + (size_t)B_ * NOUT * 4); // 8 MB

    gating_kernel<<<B_, 64, 0, stream>>>(z, c, gw0, gb0, gw1, gb1, gw2, gb2, coeff);

    const dim3 gg(NOUT / 128, B_ / 64, KSPLIT);   // (8,4,8)
    // layer 0: inp = [z | c]
    expert_gemm<<<gg, 256, 0, stream>>>(z, c, coeff, w0, partial);
    reduce_act<<<(B_ * NOUT) / 256, 256, 0, stream>>>(partial, coeff, b0, h, 1);
    // layer 1: inp = [z | h]
    expert_gemm<<<gg, 256, 0, stream>>>(z, h, coeff, w1, partial);
    reduce_act<<<(B_ * NOUT) / 256, 256, 0, stream>>>(partial, coeff, b1, h, 1);
    // layer 2: inp = [z | h], no activation
    expert_gemm<<<gg, 256, 0, stream>>>(z, h, coeff, w2, partial);
    reduce_act<<<(B_ * NOUT) / 256, 256, 0, stream>>>(partial, coeff, b2, out, 0);
}
